// MoMoShareLayer_9929964389217
// MI455X (gfx1250) — matrix operand order repaired and checked by reading, not compiled, not run on hardware
//
#include <hip/hip_runtime.h>
#include <math.h>

// ---------------------------------------------------------------------------
// MoMoShareLayer for MI455X (gfx1250): bf16 WMMA (v_wmma_f32_16x16x32_bf16)
// with f32 accumulation, sparse expert routing (top-1), gather/scatter token
// permutation so every 32-row WMMA tile uses a single expert's weights.
// Wave tile: 32x64 (2 A-frags x 4 B-frags -> 8 WMMA per K-step).
// ---------------------------------------------------------------------------

typedef __bf16 bf16_t;
typedef __attribute__((ext_vector_type(16))) __bf16 v16bf;
typedef __attribute__((ext_vector_type(8)))  float  v8f;
typedef __attribute__((ext_vector_type(8)))  unsigned int v8u;

static constexpr int BB   = 16;    // batch
static constexpr int SS   = 128;   // seq len
static constexpr int DD   = 768;   // hidden
static constexpr int FF   = 3072;  // ffn
static constexpr int UU   = 4;     // unique experts
static constexpr int EE   = 4;     // switch sub-experts
static constexpr int HH   = 12;    // heads
static constexpr int DH   = 64;    // head dim
static constexpr int NTOK = BB * SS;          // 2048
static constexpr int NP   = 2560;             // padded permuted rows (>= 2048+16*31)

// ---- WMMA fragment loaders -------------------------------------------------
// A-matrix 16x32 bf16 (ISA 7.12.2): lane l covers M=l%16; for lane half hi
// (l/16): K = (v>=4?16:0) + hi*8 + (v%4)*2 + parity.  rowp already includes
// the +hi*8 element offset.
__device__ __forceinline__ v16bf load_a_frag(const bf16_t* rowp, int koff) {
    const unsigned int* p = (const unsigned int*)(rowp + koff);
    v8u u;
    u[0] = p[0]; u[1] = p[1]; u[2] = p[2];  u[3] = p[3];
    u[4] = p[8]; u[5] = p[9]; u[6] = p[10]; u[7] = p[11];
    return __builtin_bit_cast(v16bf, u);
}

// B-matrix 32x16 bf16: lane l covers N=l%16; same K order as A:
// K = (v>=4?16:0) + hi*8 + (v%4)*2 + parity.
// rowp already includes the +(l/16)*8 element offset into the [N][K] row.
__device__ __forceinline__ v16bf load_b_frag(const bf16_t* rowp, int koff) {
    const unsigned int* p = (const unsigned int*)(rowp + koff);
    v8u u;
    u[0] = p[0]; u[1] = p[1]; u[2] = p[2];  u[3] = p[3];
    u[4] = p[8]; u[5] = p[9]; u[6] = p[10]; u[7] = p[11];
    return __builtin_bit_cast(v16bf, u);
}

// A-fragment built from an f32 row (used for softmax probs kept in LDS).
__device__ __forceinline__ v16bf frag_a_from_f32(const float* rowp, int koff) {
    v16bf r;
#pragma unroll
    for (int v = 0; v < 8; ++v) {
        int k = koff + ((v >= 4) ? 16 : 0) + (v & 3) * 2;
        r[2 * v]     = (bf16_t)rowp[k];
        r[2 * v + 1] = (bf16_t)rowp[k + 1];
    }
    return r;
}

// ---- generic WMMA GEMM: OUT(M x Nn) = A(M x K) * BT(Nn x K)^T + bias ------
// Wave tile 32x64: blockDim=128 (4 waves) covers 32 rows x 256 cols.
enum { GS_NONE = 0, GS_BATCH = 1, GS_SEG = 2 };
enum { EP_BF16 = 0, EP_VT = 1, EP_GELU_BF16 = 2, EP_F32 = 3, EP_ADDSC = 4, EP_SCATTER = 5 };

template <int GSEL, int EPI, bool AIND>
__global__ __launch_bounds__(128)
void gemm_kernel(const bf16_t* __restrict__ A, int lda,
                 const int* __restrict__ perm,
                 const bf16_t* __restrict__ BT, long wstride,
                 const float* __restrict__ bias, int bstride,
                 void* __restrict__ out,
                 int M, const int* __restrict__ mcount,
                 int Nn, int K,
                 const int* __restrict__ routes,
                 const int* __restrict__ segoff,
                 const float* __restrict__ scale) {
    const int lane = threadIdx.x & 31;
    const int wave = threadIdx.x >> 5;
    const int m0 = blockIdx.y * 32;
    const int Meff = mcount ? *mcount : M;
    if (m0 >= Meff) return;
    const int n0 = blockIdx.x * 256 + wave * 64;
    if (n0 >= Nn) return;

    int g = 0;
    if (GSEL == GS_BATCH) {
        g = routes[m0 / SS];
    } else if (GSEL == GS_SEG) {
#pragma unroll
        for (int i = 15; i >= 1; --i)
            if (m0 >= segoff[i]) { g = i; break; }
    }
    const bf16_t* BTg   = BT + (long)g * wstride;
    const float*  biasg = bias + (long)g * bstride;

    const int khiA = (lane >> 4) << 3;   // 0 or 8
    const int khiB = (lane >> 4) << 3;   // 0 or 8
    const bf16_t* arow[2];
#pragma unroll
    for (int mt = 0; mt < 2; ++mt) {
        int ri = m0 + mt * 16 + (lane & 15);
        if (AIND) { int t = perm[ri]; ri = (t < 0) ? 0 : t; }
        arow[mt] = A + (long)ri * lda + khiA;
    }
    const bf16_t* brow[4];
#pragma unroll
    for (int t = 0; t < 4; ++t)
        brow[t] = BTg + (long)(n0 + t * 16 + (lane & 15)) * K + khiB;

    const v8f vzero = {0.f, 0.f, 0.f, 0.f, 0.f, 0.f, 0.f, 0.f};
    v8f acc[2][4];
#pragma unroll
    for (int mt = 0; mt < 2; ++mt)
#pragma unroll
        for (int t = 0; t < 4; ++t) acc[mt][t] = vzero;

    for (int kb = 0; kb < K; kb += 32) {
        __builtin_prefetch(arow[0] + kb + 128, 0, 1);   // global_prefetch_b8
        v16bf a0 = load_a_frag(arow[0], kb);
        v16bf a1 = load_a_frag(arow[1], kb);
#pragma unroll
        for (int t = 0; t < 4; ++t) {
            v16bf b = load_b_frag(brow[t], kb);
            acc[0][t] = __builtin_amdgcn_wmma_f32_16x16x32_bf16(
                false, a0, false, b, (short)0, acc[0][t], false, false);
            acc[1][t] = __builtin_amdgcn_wmma_f32_16x16x32_bf16(
                false, a1, false, b, (short)0, acc[1][t], false, false);
        }
    }

    // C layout: lane l -> N = n0+t*16+(l%16), rows M = m0 + mt*16 + (l/16)*8 + r
    const int cn = lane & 15;
    const int mhalf = (lane >> 4) << 3;
#pragma unroll
    for (int mt = 0; mt < 2; ++mt) {
#pragma unroll
        for (int t = 0; t < 4; ++t) {
            const int n = n0 + t * 16 + cn;
            const float bv = biasg[n];
#pragma unroll
            for (int r = 0; r < 8; ++r) {
                const int m = m0 + mt * 16 + mhalf + r;
                float v = acc[mt][t][r] + bv;
                if (EPI == EP_BF16) {
                    ((bf16_t*)out)[(long)m * Nn + n] = (bf16_t)v;
                } else if (EPI == EP_VT) {
                    const int b_ = m / SS, s_ = m % SS, h_ = n >> 6, dh_ = n & 63;
                    ((bf16_t*)out)[(((long)(b_ * HH + h_)) * DH + dh_) * SS + s_] = (bf16_t)v;
                } else if (EPI == EP_GELU_BF16) {
                    const float gl = 0.5f * v * (1.0f + erff(v * 0.70710678118654752f));
                    ((bf16_t*)out)[(long)m * Nn + n] = (bf16_t)gl;
                } else if (EPI == EP_F32) {
                    ((float*)out)[(long)m * Nn + n] = v;
                } else if (EPI == EP_ADDSC) {
                    ((float*)out)[(long)m * Nn + n] += scale[m / SS] * v;
                } else { // EP_SCATTER
                    const int tok = perm[m];
                    if (tok >= 0)
                        ((float*)out)[(long)tok * Nn + n] += scale[tok] * v;
                }
            }
        }
    }
}

// ---- fused attention per (b,h): WMMA scores -> LDS softmax -> WMMA ctx -----
__global__ __launch_bounds__(256)
void attn_kernel(const bf16_t* __restrict__ qb, const bf16_t* __restrict__ kb,
                 const bf16_t* __restrict__ vT, const float* __restrict__ mask,
                 bf16_t* __restrict__ cxb) {
    __shared__ float sc[8][16][SS];   // 64 KB: per-wave 16x128 score rows
    const int lane = threadIdx.x & 31;
    const int wave = threadIdx.x >> 5;
    const int h = blockIdx.x, b = blockIdx.y;
    const int q0 = wave * 16;
    const int khiA = (lane >> 4) << 3;
    const int khiB = (lane >> 4) << 3;
    const int cn = lane & 15;
    const int mhalf = (lane >> 4) << 3;
    const v8f vzero = {0.f, 0.f, 0.f, 0.f, 0.f, 0.f, 0.f, 0.f};

    // scores = q @ k^T  (A rows: queries; B rows: keys; K = DH = 64)
    const bf16_t* arow = qb + ((long)(b * SS + q0 + (lane & 15))) * DD + h * DH + khiA;
    v8f acc[8];
#pragma unroll
    for (int t = 0; t < 8; ++t) acc[t] = vzero;
    for (int ks = 0; ks < DH; ks += 32) {
        v16bf a = load_a_frag(arow, ks);
#pragma unroll
        for (int t = 0; t < 8; ++t) {
            const bf16_t* brow =
                kb + ((long)(b * SS + t * 16 + (lane & 15))) * DD + h * DH + khiB;
            v16bf bb = load_b_frag(brow, ks);
            acc[t] = __builtin_amdgcn_wmma_f32_16x16x32_bf16(
                false, a, false, bb, (short)0, acc[t], false, false);
        }
    }
#pragma unroll
    for (int t = 0; t < 8; ++t) {
        const int n = t * 16 + cn;
        const float mv = mask[b * SS + n];
#pragma unroll
        for (int r = 0; r < 8; ++r)
            sc[wave][mhalf + r][n] = acc[t][r] * 0.125f + mv;   // 1/sqrt(64)
    }
    __syncthreads();

    if (lane < 16) {   // one lane per query row: softmax over 128 keys
        float* row = sc[wave][lane];
        float mx = -1e30f;
        for (int c = 0; c < SS; ++c) mx = fmaxf(mx, row[c]);
        float sum = 0.f;
        for (int c = 0; c < SS; ++c) { float e = __expf(row[c] - mx); row[c] = e; sum += e; }
        const float inv = 1.f / sum;
        for (int c = 0; c < SS; ++c) row[c] *= inv;
    }
    __syncthreads();

    // ctx = P @ V   (A rows: probs in LDS, K = SS = 128; B rows: vT[dh][s])
    const float* parow = &sc[wave][lane & 15][khiA];
    v8f cacc[4];
#pragma unroll
    for (int t = 0; t < 4; ++t) cacc[t] = vzero;
    for (int ks = 0; ks < SS; ks += 32) {
        v16bf a = frag_a_from_f32(parow, ks);
#pragma unroll
        for (int t = 0; t < 4; ++t) {
            const bf16_t* brow =
                vT + (((long)(b * HH + h)) * DH + t * 16 + (lane & 15)) * SS + khiB;
            v16bf bb = load_b_frag(brow, ks);
            cacc[t] = __builtin_amdgcn_wmma_f32_16x16x32_bf16(
                false, a, false, bb, (short)0, cacc[t], false, false);
        }
    }
#pragma unroll
    for (int t = 0; t < 4; ++t) {
        const int n = t * 16 + cn;
#pragma unroll
        for (int r = 0; r < 8; ++r)
            cxb[((long)(b * SS + q0 + mhalf + r)) * DD + h * DH + n] = (bf16_t)cacc[t][r];
    }
}

// ---- small helper kernels --------------------------------------------------
__global__ __launch_bounds__(256)
void transpose_cvt_kernel(const float* __restrict__ in, bf16_t* __restrict__ out,
                          int K, int Nn) {
    __shared__ float tile[32][33];
    const long g = blockIdx.z;
    const float* inp = in + g * (long)K * Nn;
    bf16_t* outp = out + g * (long)K * Nn;
    const int k0 = blockIdx.y * 32, n0 = blockIdx.x * 32;
    const int tx = threadIdx.x & 31, ty = threadIdx.x >> 5;
    for (int i = 0; i < 32; i += 8) {
        const int k = k0 + ty + i, n = n0 + tx;
        if (k < K && n < Nn) tile[ty + i][tx] = inp[(long)k * Nn + n];
    }
    __syncthreads();
    for (int i = 0; i < 32; i += 8) {
        const int n = n0 + ty + i, k = k0 + tx;
        if (n < Nn && k < K) outp[(long)n * K + k] = (bf16_t)tile[tx][ty + i];
    }
}

__global__ void cvt_bf16_kernel(const float* __restrict__ in,
                                bf16_t* __restrict__ out, long n) {
    const long i = (long)blockIdx.x * 256 + threadIdx.x;
    if (i < n) out[i] = (bf16_t)in[i];
}

__global__ __launch_bounds__(256)
void batch_route_kernel(const float* __restrict__ x, const float* __restrict__ sw_w,
                        const float* __restrict__ sw_b, int* __restrict__ routes,
                        float* __restrict__ pmax) {
    __shared__ float red[256][4];
    const int b = blockIdx.x, t = threadIdx.x;
    float l[4] = {0.f, 0.f, 0.f, 0.f};
    for (int d = t; d < DD; d += 256) {
        float s = 0.f;
        for (int s_ = 0; s_ < SS; ++s_) s += x[((long)b * SS + s_) * DD + d];
        s *= (1.0f / SS);
#pragma unroll
        for (int u = 0; u < UU; ++u) l[u] += s * sw_w[d * UU + u];
    }
#pragma unroll
    for (int u = 0; u < 4; ++u) red[t][u] = l[u];
    __syncthreads();
    for (int off = 128; off > 0; off >>= 1) {
        if (t < off)
            for (int u = 0; u < 4; ++u) red[t][u] += red[t + off][u];
        __syncthreads();
    }
    if (t == 0) {
        float lg[4], mx = -1e30f;
        for (int u = 0; u < 4; ++u) { lg[u] = red[0][u] + sw_b[u]; mx = fmaxf(mx, lg[u]); }
        float sum = 0.f;
        for (int u = 0; u < 4; ++u) { lg[u] = __expf(lg[u] - mx); sum += lg[u]; }
        int am = 0; float pm = lg[0];
        for (int u = 1; u < 4; ++u) if (lg[u] > pm) { pm = lg[u]; am = u; }
        routes[b] = am; pmax[b] = pm / sum;
    }
}

__global__ __launch_bounds__(256)
void tok_route_kernel(const float* __restrict__ att, const float* __restrict__ ur_w,
                      const float* __restrict__ ur_b, const int* __restrict__ routes,
                      int* __restrict__ tg, float* __restrict__ tp,
                      int* __restrict__ counts) {
    const int lane = threadIdx.x & 31, wave = threadIdx.x >> 5;
    const int tok = blockIdx.x * 8 + wave;
    if (tok >= NTOK) return;
    const int u = routes[tok / SS];
    const float* wr = ur_w + (long)u * DD * EE;
    float l[4] = {0.f, 0.f, 0.f, 0.f};
    for (int d = lane; d < DD; d += 32) {
        const float xv = att[(long)tok * DD + d];
#pragma unroll
        for (int e = 0; e < EE; ++e) l[e] += xv * wr[d * EE + e];
    }
#pragma unroll
    for (int off = 16; off > 0; off >>= 1) {
#pragma unroll
        for (int e = 0; e < 4; ++e) l[e] += __shfl_down(l[e], off, 32);
    }
    if (lane == 0) {
        float mx = -1e30f;
        for (int e = 0; e < 4; ++e) { l[e] += ur_b[u * EE + e]; mx = fmaxf(mx, l[e]); }
        float sum = 0.f;
        for (int e = 0; e < 4; ++e) { l[e] = __expf(l[e] - mx); sum += l[e]; }
        int am = 0; float pm = l[0];
        for (int e = 1; e < 4; ++e) if (l[e] > pm) { pm = l[e]; am = e; }
        const int g = u * EE + am;
        tg[tok] = g; tp[tok] = pm / sum;
        atomicAdd(&counts[g], 1);
    }
}

__global__ void init_route_kernel(int* counts, int* cursor, int* perm, int np) {
    const int i = blockIdx.x * 256 + threadIdx.x;
    if (i < 16) { counts[i] = 0; cursor[i] = 0; }
    if (i < np) perm[i] = -1;
}

__global__ void seg_scan_kernel(const int* counts, int* segoff, int* mcount) {
    if (threadIdx.x == 0 && blockIdx.x == 0) {
        int off = 0;
        for (int g = 0; g < 16; ++g) { segoff[g] = off; off += (counts[g] + 31) & ~31; }
        segoff[16] = off;
        mcount[0] = off;
    }
}

__global__ void fill_perm_kernel(const int* tg, const int* segoff, int* cursor,
                                 int* perm) {
    const int tok = blockIdx.x * 256 + threadIdx.x;
    if (tok >= NTOK) return;
    const int g = tg[tok];
    const int pos = segoff[g] + atomicAdd(&cursor[g], 1);
    perm[pos] = tok;
}

__global__ __launch_bounds__(256)
void layernorm_kernel(const float* __restrict__ att, const float* __restrict__ ffn,
                      const float* __restrict__ gamma, const float* __restrict__ beta,
                      float* __restrict__ out) {
    __shared__ float rs[256], rs2[256];
    const int tok = blockIdx.x, t = threadIdx.x;
    float s = 0.f, s2 = 0.f, v[3];
#pragma unroll
    for (int i = 0; i < 3; ++i) {
        const int d = t + i * 256;
        const float y = att[(long)tok * DD + d] + ffn[(long)tok * DD + d];
        v[i] = y; s += y; s2 += y * y;
    }
    rs[t] = s; rs2[t] = s2;
    __syncthreads();
    for (int off = 128; off > 0; off >>= 1) {
        if (t < off) { rs[t] += rs[t + off]; rs2[t] += rs2[t + off]; }
        __syncthreads();
    }
    const float mean = rs[0] * (1.f / DD);
    const float var = rs2[0] * (1.f / DD) - mean * mean;
    const float inv = rsqrtf(var + 1e-12f);
#pragma unroll
    for (int i = 0; i < 3; ++i) {
        const int d = t + i * 256;
        out[(long)tok * DD + d] = (v[i] - mean) * inv * gamma[d] + beta[d];
    }
}

// ---------------------------------------------------------------------------
extern "C" void kernel_launch(void* const* d_in, const int* in_sizes, int n_in,
                              void* d_out, int out_size, void* d_ws, size_t ws_size,
                              hipStream_t stream) {
    (void)in_sizes; (void)n_in; (void)out_size; (void)ws_size;
    const float* x     = (const float*)d_in[0];
    const float* mask  = (const float*)d_in[1];
    const float* sw_w  = (const float*)d_in[2];
    const float* sw_b  = (const float*)d_in[3];
    const float* ca_wq = (const float*)d_in[4];  const float* ca_bq = (const float*)d_in[5];
    const float* ca_wk = (const float*)d_in[6];  const float* ca_bk = (const float*)d_in[7];
    const float* ca_wv = (const float*)d_in[8];  const float* ca_bv = (const float*)d_in[9];
    const float* ca_wo = (const float*)d_in[10]; const float* ca_bo = (const float*)d_in[11];
    const float* ua_wq = (const float*)d_in[12]; const float* ua_bq = (const float*)d_in[13];
    const float* ua_wk = (const float*)d_in[14]; const float* ua_bk = (const float*)d_in[15];
    const float* ua_wv = (const float*)d_in[16]; const float* ua_bv = (const float*)d_in[17];
    const float* ua_wo = (const float*)d_in[18]; const float* ua_bo = (const float*)d_in[19];
    const float* cf_w1 = (const float*)d_in[20]; const float* cf_b1 = (const float*)d_in[21];
    const float* cf_w2 = (const float*)d_in[22]; const float* cf_b2 = (const float*)d_in[23];
    const float* ur_w  = (const float*)d_in[24]; const float* ur_b  = (const float*)d_in[25];
    const float* uf_w1 = (const float*)d_in[26]; const float* uf_b1 = (const float*)d_in[27];
    const float* uf_w2 = (const float*)d_in[28]; const float* uf_b2 = (const float*)d_in[29];
    const float* ln_g  = (const float*)d_in[30]; const float* ln_b  = (const float*)d_in[31];
    float* out = (float*)d_out;

    char* wsp = (char*)d_ws;
    auto take = [&](size_t bytes) -> void* {
        void* r = (void*)wsp;
        wsp += (bytes + 255) & ~(size_t)255;
        return r;
    };
    bf16_t* xb   = (bf16_t*)take((size_t)NTOK * DD * 2);
    bf16_t* qb   = (bf16_t*)take((size_t)NTOK * DD * 2);
    bf16_t* kb   = (bf16_t*)take((size_t)NTOK * DD * 2);
    bf16_t* vT   = (bf16_t*)take((size_t)NTOK * DD * 2);
    bf16_t* cxb  = (bf16_t*)take((size_t)NTOK * DD * 2);
    bf16_t* attb = (bf16_t*)take((size_t)NTOK * DD * 2);
    float*  att  = (float*)take((size_t)NTOK * DD * 4);
    float*  ffn  = (float*)take((size_t)NTOK * DD * 4);
    bf16_t* Hc   = (bf16_t*)take((size_t)NTOK * FF * 2);
    bf16_t* Hu   = (bf16_t*)take((size_t)NP * FF * 2);
    int*   routes = (int*)take(BB * 4);
    float* pmax   = (float*)take(BB * 4);
    int*   tg     = (int*)take(NTOK * 4);
    float* tp     = (float*)take(NTOK * 4);
    int*   counts = (int*)take(64);
    int*   cursor = (int*)take(64);
    int*   segoff = (int*)take(32 * 4);
    int*   mcount = (int*)take(64);
    int*   perm   = (int*)take(NP * 4);
    bf16_t* cqT  = (bf16_t*)take((size_t)DD * DD * 2);
    bf16_t* ckT  = (bf16_t*)take((size_t)DD * DD * 2);
    bf16_t* cvT  = (bf16_t*)take((size_t)DD * DD * 2);
    bf16_t* coT  = (bf16_t*)take((size_t)DD * DD * 2);
    bf16_t* uqT  = (bf16_t*)take((size_t)UU * DD * DD * 2);
    bf16_t* ukT  = (bf16_t*)take((size_t)UU * DD * DD * 2);
    bf16_t* uvT  = (bf16_t*)take((size_t)UU * DD * DD * 2);
    bf16_t* uoT  = (bf16_t*)take((size_t)UU * DD * DD * 2);
    bf16_t* cf1T = (bf16_t*)take((size_t)DD * FF * 2);
    bf16_t* cf2T = (bf16_t*)take((size_t)DD * FF * 2);
    bf16_t* uf1T = (bf16_t*)take((size_t)UU * EE * DD * FF * 2);
    bf16_t* uf2T = (bf16_t*)take((size_t)UU * EE * DD * FF * 2);

    // 1) batch routing + x -> bf16
    batch_route_kernel<<<BB, 256, 0, stream>>>(x, sw_w, sw_b, routes, pmax);
    const long nx = (long)NTOK * DD;
    cvt_bf16_kernel<<<(unsigned)((nx + 255) / 256), 256, 0, stream>>>(x, xb, nx);

    // 2) weight transpose+convert to bf16 [N][K]
    transpose_cvt_kernel<<<dim3(DD / 32, DD / 32, 1), 256, 0, stream>>>(ca_wq, cqT, DD, DD);
    transpose_cvt_kernel<<<dim3(DD / 32, DD / 32, 1), 256, 0, stream>>>(ca_wk, ckT, DD, DD);
    transpose_cvt_kernel<<<dim3(DD / 32, DD / 32, 1), 256, 0, stream>>>(ca_wv, cvT, DD, DD);
    transpose_cvt_kernel<<<dim3(DD / 32, DD / 32, 1), 256, 0, stream>>>(ca_wo, coT, DD, DD);
    transpose_cvt_kernel<<<dim3(DD / 32, DD / 32, UU), 256, 0, stream>>>(ua_wq, uqT, DD, DD);
    transpose_cvt_kernel<<<dim3(DD / 32, DD / 32, UU), 256, 0, stream>>>(ua_wk, ukT, DD, DD);
    transpose_cvt_kernel<<<dim3(DD / 32, DD / 32, UU), 256, 0, stream>>>(ua_wv, uvT, DD, DD);
    transpose_cvt_kernel<<<dim3(DD / 32, DD / 32, UU), 256, 0, stream>>>(ua_wo, uoT, DD, DD);
    transpose_cvt_kernel<<<dim3(FF / 32, DD / 32, 1), 256, 0, stream>>>(cf_w1, cf1T, DD, FF);
    transpose_cvt_kernel<<<dim3(DD / 32, FF / 32, 1), 256, 0, stream>>>(cf_w2, cf2T, FF, DD);
    transpose_cvt_kernel<<<dim3(FF / 32, DD / 32, UU * EE), 256, 0, stream>>>(uf_w1, uf1T, DD, FF);
    transpose_cvt_kernel<<<dim3(DD / 32, FF / 32, UU * EE), 256, 0, stream>>>(uf_w2, uf2T, FF, DD);

    const dim3 gD(DD / 256, NTOK / 32);   // (3,64)
    const dim3 gF(FF / 256, NTOK / 32);   // (12,64)

    // 3) common attention
    gemm_kernel<GS_NONE, EP_BF16, false><<<gD, 128, 0, stream>>>(
        xb, DD, nullptr, cqT, 0, ca_bq, 0, qb, NTOK, nullptr, DD, DD, nullptr, nullptr, nullptr);
    gemm_kernel<GS_NONE, EP_BF16, false><<<gD, 128, 0, stream>>>(
        xb, DD, nullptr, ckT, 0, ca_bk, 0, kb, NTOK, nullptr, DD, DD, nullptr, nullptr, nullptr);
    gemm_kernel<GS_NONE, EP_VT, false><<<gD, 128, 0, stream>>>(
        xb, DD, nullptr, cvT, 0, ca_bv, 0, vT, NTOK, nullptr, DD, DD, nullptr, nullptr, nullptr);
    attn_kernel<<<dim3(HH, BB), 256, 0, stream>>>(qb, kb, vT, mask, cxb);
    gemm_kernel<GS_NONE, EP_F32, false><<<gD, 128, 0, stream>>>(
        cxb, DD, nullptr, coT, 0, ca_bo, 0, att, NTOK, nullptr, DD, DD, nullptr, nullptr, nullptr);

    // 4) unique attention (only the routed expert per batch row)
    gemm_kernel<GS_BATCH, EP_BF16, false><<<gD, 128, 0, stream>>>(
        xb, DD, nullptr, uqT, (long)DD * DD, ua_bq, DD, qb, NTOK, nullptr, DD, DD, routes, nullptr, nullptr);
    gemm_kernel<GS_BATCH, EP_BF16, false><<<gD, 128, 0, stream>>>(
        xb, DD, nullptr, ukT, (long)DD * DD, ua_bk, DD, kb, NTOK, nullptr, DD, DD, routes, nullptr, nullptr);
    gemm_kernel<GS_BATCH, EP_VT, false><<<gD, 128, 0, stream>>>(
        xb, DD, nullptr, uvT, (long)DD * DD, ua_bv, DD, vT, NTOK, nullptr, DD, DD, routes, nullptr, nullptr);
    attn_kernel<<<dim3(HH, BB), 256, 0, stream>>>(qb, kb, vT, mask, cxb);
    gemm_kernel<GS_BATCH, EP_ADDSC, false><<<gD, 128, 0, stream>>>(
        cxb, DD, nullptr, uoT, (long)DD * DD, ua_bo, DD, att, NTOK, nullptr, DD, DD, routes, nullptr, pmax);

    // 5) token-level switch routing + permutation (segments padded to 32 rows)
    cvt_bf16_kernel<<<(unsigned)((nx + 255) / 256), 256, 0, stream>>>(att, attb, nx);
    init_route_kernel<<<(NP + 255) / 256, 256, 0, stream>>>(counts, cursor, perm, NP);
    tok_route_kernel<<<NTOK / 8, 256, 0, stream>>>(att, ur_w, ur_b, routes, tg, tp, counts);
    seg_scan_kernel<<<1, 32, 0, stream>>>(counts, segoff, mcount);
    fill_perm_kernel<<<NTOK / 256, 256, 0, stream>>>(tg, segoff, cursor, perm);

    // 6) common FFN
    gemm_kernel<GS_NONE, EP_GELU_BF16, false><<<gF, 128, 0, stream>>>(
        attb, DD, nullptr, cf1T, 0, cf_b1, 0, Hc, NTOK, nullptr, FF, DD, nullptr, nullptr, nullptr);
    gemm_kernel<GS_NONE, EP_F32, false><<<gD, 128, 0, stream>>>(
        Hc, FF, nullptr, cf2T, 0, cf_b2, 0, ffn, NTOK, nullptr, DD, FF, nullptr, nullptr, nullptr);

    // 7) unique switch FFN over gathered, expert-uniform 32-row tiles
    gemm_kernel<GS_SEG, EP_GELU_BF16, true><<<dim3(FF / 256, NP / 32), 128, 0, stream>>>(
        attb, DD, perm, uf1T, (long)DD * FF, uf_b1, FF, Hu, NP, mcount, FF, DD, nullptr, segoff, nullptr);
    gemm_kernel<GS_SEG, EP_SCATTER, false><<<dim3(DD / 256, NP / 32), 128, 0, stream>>>(
        Hu, FF, perm, uf2T, (long)DD * FF, uf_b2, DD, ffn, NP, mcount, DD, FF, nullptr, segoff, tp);

    // 8) residual + layernorm
    layernorm_kernel<<<NTOK, 256, 0, stream>>>(att, ffn, ln_g, ln_b, out);
}
